// wRNN_50706383896679
// MI455X (gfx1250) — matrix operand order repaired and checked by reading, not compiled, not run on hardware
//
#include <hip/hip_runtime.h>
#include <hip/hip_bf16.h>
#include <math.h>

// MI455X / gfx1250: wave32, WMMA 16x16x32 bf16 with f32 accumulation.
// Weights (~44MB as bf16 shadows) are L2-resident (192MB L2); the T=256
// sequential scan runs as a chain of kernels on `stream`.
// Logit GEMM stages its 64x1024 A-matrix into LDS with async global->LDS
// copies (ASYNCcnt) and reads fragments back via ds_load_b128.

typedef __attribute__((ext_vector_type(16))) __bf16 v16bf;
typedef __attribute__((ext_vector_type(8)))  float  v8f;
typedef unsigned short u16;
typedef unsigned int   u32;

#define TPB 256
#define WAVES_PER_BLOCK 8

static __device__ __forceinline__ u16 f2bf(float f) {
  union { float f; u32 u; } v; v.f = f;
  u32 r = v.u + 0x7FFFu + ((v.u >> 16) & 1u);   // round-to-nearest-even
  return (u16)(r >> 16);
}

static __device__ __forceinline__ float blk_reduce_sum(float v, float* red) {
  red[threadIdx.x] = v;
  __syncthreads();
  for (int st = TPB / 2; st > 0; st >>= 1) {
    if ((int)threadIdx.x < st) red[threadIdx.x] += red[threadIdx.x + st];
    __syncthreads();
  }
  float r = red[0];
  __syncthreads();
  return r;
}

static __device__ __forceinline__ float blk_reduce_max(float v, float* red) {
  red[threadIdx.x] = v;
  __syncthreads();
  for (int st = TPB / 2; st > 0; st >>= 1) {
    if ((int)threadIdx.x < st)
      red[threadIdx.x] = fmaxf(red[threadIdx.x], red[threadIdx.x + st]);
    __syncthreads();
  }
  float r = red[0];
  __syncthreads();
  return r;
}

// ---------------------------------------------------------------------------
// Generic: D[32,N] = [silu]( A[32,K](bf16) @ W[N,K]^T(bf16) + bias ) [+ accin]
// One wave -> one 16x16 tile via v_wmma_f32_16x16x32_bf16, K-loop step 32.
// Fragment layouts per CDNA5 ISA 7.12.2.
// ---------------------------------------------------------------------------
template <int DO_SILU, int DO_ACC>
__global__ __launch_bounds__(TPB)
void gemm_nt(const u16* __restrict__ A, const u16* __restrict__ W,
             const float* __restrict__ bias, const float* __restrict__ accin,
             float* __restrict__ outf, u16* __restrict__ outb,
             int K, int N, long long ldout) {
  const int lane = threadIdx.x & 31;
  const int wave = threadIdx.x >> 5;
  const int tile = blockIdx.x * WAVES_PER_BLOCK + wave;
  const int nTilesN = N >> 4;
  if (tile >= 2 * nTilesN) return;        // wave-uniform exit (EXEC stays full)
  const int tm = tile & 1;
  const int tn = tile >> 1;
  const int half = lane >> 4;
  const int l16 = lane & 15;

  const u16* arow = A + (size_t)(tm * 16 + l16) * K;
  const u16* wrow = W + (size_t)(tn * 16 + l16) * K;
  const int aoff0 = half ? 8 : 0, aoff1 = half ? 24 : 16;
  const int boff0 = half ? 8 : 0, boff1 = half ? 24 : 16;

  v8f c = {};
  for (int k0 = 0; k0 < K; k0 += 32) {
    v16bf af, bfv;
    uint4* ap = reinterpret_cast<uint4*>(&af);
    uint4* bp = reinterpret_cast<uint4*>(&bfv);
    ap[0] = *reinterpret_cast<const uint4*>(arow + k0 + aoff0);
    ap[1] = *reinterpret_cast<const uint4*>(arow + k0 + aoff1);
    bp[0] = *reinterpret_cast<const uint4*>(wrow + k0 + boff0);
    bp[1] = *reinterpret_cast<const uint4*>(wrow + k0 + boff1);
    __builtin_prefetch(wrow + k0 + 128, 0, 3);  // high-temporal: keep in near caches
    c = __builtin_amdgcn_wmma_f32_16x16x32_bf16(false, af, false, bfv,
                                                (short)0, c, false, false);
  }

  const int n = tn * 16 + l16;
  const float bv = bias ? bias[n] : 0.0f;
#pragma unroll
  for (int j = 0; j < 8; ++j) {
    const int m = tm * 16 + (half ? 8 : 0) + j;
    float v = c[j] + bv;
    if (DO_SILU) v = v / (1.0f + __expf(-v));
    if (DO_ACC) v += accin[(size_t)m * N + n];
    if (outf) outf[(size_t)m * (size_t)ldout + n] = v;
    if (outb) outb[(size_t)m * N + n] = f2bf(v);
  }
}

// ---------------------------------------------------------------------------
// Fused head GEMM: A = xn[32,E]; W = [cur_w ; nxt_w] as [2E,E] bf16.
//   n <  E : cur_emb[m,n] = v ; emb_cat[m,n]      = bf16(v)
//   n >= E : state[m,n-E] = v ; emb_cat[32+m,n-E] = bf16(v)   (nxt_emb)
// ---------------------------------------------------------------------------
__global__ __launch_bounds__(TPB)
void head_gemm(const u16* __restrict__ A, const u16* __restrict__ W,
               const float* __restrict__ curb, const float* __restrict__ nxtb,
               float* __restrict__ cur_emb, float* __restrict__ state,
               u16* __restrict__ emb_cat, int E) {
  const int K = E, N = 2 * E;
  const int lane = threadIdx.x & 31;
  const int wave = threadIdx.x >> 5;
  const int tile = blockIdx.x * WAVES_PER_BLOCK + wave;
  const int nTilesN = N >> 4;
  if (tile >= 2 * nTilesN) return;
  const int tm = tile & 1;
  const int tn = tile >> 1;
  const int half = lane >> 4;
  const int l16 = lane & 15;

  const u16* arow = A + (size_t)(tm * 16 + l16) * K;
  const u16* wrow = W + (size_t)(tn * 16 + l16) * K;
  const int aoff0 = half ? 8 : 0, aoff1 = half ? 24 : 16;
  const int boff0 = half ? 8 : 0, boff1 = half ? 24 : 16;

  v8f c = {};
  for (int k0 = 0; k0 < K; k0 += 32) {
    v16bf af, bfv;
    uint4* ap = reinterpret_cast<uint4*>(&af);
    uint4* bp = reinterpret_cast<uint4*>(&bfv);
    ap[0] = *reinterpret_cast<const uint4*>(arow + k0 + aoff0);
    ap[1] = *reinterpret_cast<const uint4*>(arow + k0 + aoff1);
    bp[0] = *reinterpret_cast<const uint4*>(wrow + k0 + boff0);
    bp[1] = *reinterpret_cast<const uint4*>(wrow + k0 + boff1);
    __builtin_prefetch(wrow + k0 + 128, 0, 3);
    c = __builtin_amdgcn_wmma_f32_16x16x32_bf16(false, af, false, bfv,
                                                (short)0, c, false, false);
  }

  const int n = tn * 16 + l16;
  const bool isCur = (n < E);
  const int nc = isCur ? n : n - E;
  const float bv = isCur ? curb[nc] : nxtb[nc];
#pragma unroll
  for (int j = 0; j < 8; ++j) {
    const int m = tm * 16 + (half ? 8 : 0) + j;
    float v = c[j] + bv;
    if (isCur) {
      cur_emb[(size_t)m * E + nc] = v;
      emb_cat[(size_t)m * E + nc] = f2bf(v);
    } else {
      state[(size_t)m * E + nc] = v;
      emb_cat[(size_t)(32 + m) * E + nc] = f2bf(v);
    }
  }
}

// ---------------------------------------------------------------------------
// Fused logits GEMM: A_cat[64,E] bf16 (rows 0-31 cur_emb, 32-63 nxt_emb),
// W = emb table [V,E] bf16.  A is staged into LDS (128KB dynamic) with
// global_load_async_to_lds_b128 (ASYNCcnt), then fragments read via ds_load.
//   m <  32 : cur_logit[m,n]
//   m >= 32 : out_t[(m-32)*out_ld + n]   (nxt logits -> d_out)
// ---------------------------------------------------------------------------
__global__ __launch_bounds__(TPB)
void logit_gemm(const u16* __restrict__ Acat, const u16* __restrict__ embw,
                float* __restrict__ cur_logit, float* __restrict__ out_t,
                long long out_ld, int E, int V) {
  extern __shared__ u16 As[];  // 64 * E bf16 = 128KB (WGP has 320KB LDS)

  // Cooperative async stage: 64*E*2 bytes as b128 transfers.
  {
    const u32 ldsbase = (u32)(size_t)&As[0];  // low 32 bits of flat LDS addr
    const int nxfer = (64 * E) / 8;           // 16B per transfer
    for (int i = threadIdx.x; i < nxfer; i += TPB) {
      const u16* g = Acat + (size_t)i * 8;
      const u32 l = ldsbase + (u32)i * 16;
      asm volatile("global_load_async_to_lds_b128 %0, %1, off"
                   :: "v"(l), "v"(g) : "memory");
    }
    asm volatile("s_wait_asynccnt 0x0" ::: "memory");
  }
  __syncthreads();

  const int lane = threadIdx.x & 31;
  const int wave = threadIdx.x >> 5;
  const int tile = blockIdx.x * WAVES_PER_BLOCK + wave;
  const int nTilesN = V >> 4;
  if (tile >= 4 * nTilesN) return;
  const int tm = tile & 3;       // 4 row tiles (64 rows)
  const int tn = tile >> 2;
  const int half = lane >> 4;
  const int l16 = lane & 15;

  const u16* arow = As + (size_t)(tm * 16 + l16) * E;    // LDS reads
  const u16* wrow = embw + (size_t)(tn * 16 + l16) * E;  // global (L2) reads
  const int aoff0 = half ? 8 : 0, aoff1 = half ? 24 : 16;
  const int boff0 = half ? 8 : 0, boff1 = half ? 24 : 16;

  v8f c = {};
  for (int k0 = 0; k0 < E; k0 += 32) {
    v16bf af, bfv;
    uint4* ap = reinterpret_cast<uint4*>(&af);
    uint4* bp = reinterpret_cast<uint4*>(&bfv);
    ap[0] = *reinterpret_cast<const uint4*>(arow + k0 + aoff0);  // ds_load_b128
    ap[1] = *reinterpret_cast<const uint4*>(arow + k0 + aoff1);
    bp[0] = *reinterpret_cast<const uint4*>(wrow + k0 + boff0);
    bp[1] = *reinterpret_cast<const uint4*>(wrow + k0 + boff1);
    __builtin_prefetch(wrow + k0 + 128, 0, 3);
    c = __builtin_amdgcn_wmma_f32_16x16x32_bf16(false, af, false, bfv,
                                                (short)0, c, false, false);
  }

  const int n = tn * 16 + l16;
#pragma unroll
  for (int j = 0; j < 8; ++j) {
    const int m = tm * 16 + (half ? 8 : 0) + j;
    const float v = c[j];
    if (m < 32) cur_logit[(size_t)m * V + n] = v;
    else        out_t[(size_t)(m - 32) * (size_t)out_ld + n] = v;
  }
}

// out_bf16[row,:] = (a[row,:] (+ b2[row,:])) * rsqrt(mean(sq)+eps) * w[:]
__global__ __launch_bounds__(TPB)
void rms_bf16(const float* __restrict__ a, const float* __restrict__ b2,
              const float* __restrict__ w, u16* __restrict__ outb, int E) {
  __shared__ float red[TPB];
  const int row = blockIdx.x;
  const float* pa = a + (size_t)row * E;
  const float* pb = b2 ? b2 + (size_t)row * E : nullptr;
  float s = 0.0f;
  for (int e = threadIdx.x; e < E; e += TPB) {
    float v = pa[e] + (pb ? pb[e] : 0.0f);
    s += v * v;
  }
  float tot = blk_reduce_sum(s, red);
  const float scale = rsqrtf(tot / (float)E + 1.1920929e-07f);
  for (int e = threadIdx.x; e < E; e += TPB) {
    float v = pa[e] + (pb ? pb[e] : 0.0f);
    outb[(size_t)row * E + e] = f2bf(v * scale * w[e]);
  }
}

// inp = emb_w[tokens[:,t]] + state ; h = inp
__global__ __launch_bounds__(TPB)
void step_init(const int* __restrict__ tokens, int t, int T,
               const float* __restrict__ emb, const float* __restrict__ state,
               float* __restrict__ inp, float* __restrict__ h, int E) {
  const int b = blockIdx.y;
  const int e = blockIdx.x * TPB + threadIdx.x;
  const int tok = tokens[b * T + t];
  const float v = emb[(size_t)tok * E + e] + state[(size_t)b * E + e];
  inp[(size_t)b * E + e] = v;
  h[(size_t)b * E + e] = v;
}

__global__ __launch_bounds__(TPB)
void aux_kernel(const float* __restrict__ cur_emb, const float* __restrict__ inp,
                float* __restrict__ auxbuf, int t, int n) {
  __shared__ float red[TPB];
  float s = 0.0f;
  for (int i = threadIdx.x; i < n; i += TPB) {
    float d = cur_emb[i] - inp[i];
    s += d * d;
  }
  float tot = blk_reduce_sum(s, red);
  if (threadIdx.x == 0) auxbuf[t] = tot / (float)n;
}

// -log p(target) for one (row, cur/nxt) pair
__global__ __launch_bounds__(TPB)
void ce_kernel(const float* __restrict__ cur_logit, const float* __restrict__ nxt_base,
               long long nxt_ld, const int* __restrict__ tokens,
               const int* __restrict__ targets, int t, int T, int V,
               float* __restrict__ ce_rows) {
  __shared__ float red[TPB];
  const int b = blockIdx.x, which = blockIdx.y;
  const float* row = (which == 0) ? (cur_logit + (size_t)b * V)
                                  : (nxt_base + (size_t)b * (size_t)nxt_ld);
  const int tgt = (which == 0) ? tokens[b * T + t] : targets[b * T + t];
  float m = -1e30f;
  for (int i = threadIdx.x; i < V; i += TPB) m = fmaxf(m, row[i]);
  float M = blk_reduce_max(m, red);
  float s = 0.0f;
  for (int i = threadIdx.x; i < V; i += TPB) s += __expf(row[i] - M);
  float S = blk_reduce_sum(s, red);
  if (threadIdx.x == 0)
    ce_rows[(size_t)t * 64 + which * 32 + b] = -(row[tgt] - M - __logf(S));
}

__global__ __launch_bounds__(TPB)
void final_reduce(const float* __restrict__ ce_rows, const float* __restrict__ auxbuf,
                  float* __restrict__ out2) {
  __shared__ float red[TPB];
  float ce = 0.0f, ax = 0.0f;
  for (int t = threadIdx.x; t < 256; t += TPB) {
    float sc = 0.0f, sn = 0.0f;
    for (int b = 0; b < 32; ++b) {
      sc += ce_rows[t * 64 + b];
      sn += ce_rows[t * 64 + 32 + b];
    }
    ce += 0.5f * (sc + sn) / 32.0f;
    ax += auxbuf[t];
  }
  float ces = blk_reduce_sum(ce, red);
  float axs = blk_reduce_sum(ax, red);
  if (threadIdx.x == 0) {
    out2[0] = ces / 256.0f;
    out2[1] = axs / 256.0f;
  }
}

__global__ __launch_bounds__(TPB)
void to_bf16_k(const float* __restrict__ in, u16* __restrict__ out, size_t n) {
  size_t i = (size_t)blockIdx.x * TPB + threadIdx.x;
  if (i < n) out[i] = f2bf(in[i]);
}

__global__ __launch_bounds__(TPB)
void zero_f32(float* __restrict__ p, int n) {
  int i = blockIdx.x * TPB + threadIdx.x;
  if (i < n) p[i] = 0.0f;
}

extern "C" void kernel_launch(void* const* d_in, const int* in_sizes, int n_in,
                              void* d_out, int out_size, void* d_ws, size_t ws_size,
                              hipStream_t stream) {
  const int B = 32, T = 256, E = 1024, M = 4096, V = 4096, L = 2;
  const int* tokens  = (const int*)d_in[0];
  const int* targets = (const int*)d_in[1];
  const float* emb_w = (const float*)d_in[2];
  const float* bnw   = (const float*)d_in[3];
  const float* fc1w  = (const float*)d_in[4];
  const float* fc1b  = (const float*)d_in[5];
  const float* fc2w  = (const float*)d_in[6];
  const float* fc2b  = (const float*)d_in[7];
  const float* normw = (const float*)d_in[8];
  const float* curw  = (const float*)d_in[9];
  const float* curb  = (const float*)d_in[10];
  const float* nxtw  = (const float*)d_in[11];
  const float* nxtb  = (const float*)d_in[12];
  float* out = (float*)d_out;

  // Workspace carve-up (~46MB; bf16 weight shadows stay L2-resident)
  char* p = (char*)d_ws;
  auto carve = [&](size_t bytes) {
    char* r = p;
    p += (bytes + 255) & ~(size_t)255;
    return r;
  };
  u16* emb_bf   = (u16*)carve((size_t)V * E * 2);
  u16* fc1_bf   = (u16*)carve((size_t)L * M * E * 2);
  u16* fc2_bf   = (u16*)carve((size_t)L * E * M * 2);
  u16* heads_bf = (u16*)carve((size_t)2 * E * E * 2);  // [cur_w ; nxt_w]
  float* state   = (float*)carve((size_t)B * E * 4);
  float* inp     = (float*)carve((size_t)B * E * 4);
  float* h       = (float*)carve((size_t)B * E * 4);
  float* cur_emb = (float*)carve((size_t)B * E * 4);
  u16* xn_bf      = (u16*)carve((size_t)B * E * 2);
  u16* y_bf       = (u16*)carve((size_t)B * M * 2);
  u16* emb_cat_bf = (u16*)carve((size_t)2 * B * E * 2);  // [cur_emb ; nxt_emb]
  float* cur_logit = (float*)carve((size_t)B * V * 4);
  float* ce_rows   = (float*)carve((size_t)T * 64 * 4);
  float* auxbuf    = (float*)carve((size_t)T * 4);

  auto gblocks = [](size_t n) { return (unsigned)((n + TPB - 1) / TPB); };
  to_bf16_k<<<gblocks((size_t)V * E), TPB, 0, stream>>>(emb_w, emb_bf, (size_t)V * E);
  to_bf16_k<<<gblocks((size_t)L * M * E), TPB, 0, stream>>>(fc1w, fc1_bf, (size_t)L * M * E);
  to_bf16_k<<<gblocks((size_t)L * E * M), TPB, 0, stream>>>(fc2w, fc2_bf, (size_t)L * E * M);
  to_bf16_k<<<gblocks((size_t)E * E), TPB, 0, stream>>>(curw, heads_bf, (size_t)E * E);
  to_bf16_k<<<gblocks((size_t)E * E), TPB, 0, stream>>>(nxtw, heads_bf + (size_t)E * E,
                                                        (size_t)E * E);
  zero_f32<<<gblocks((size_t)B * E), TPB, 0, stream>>>(state, B * E);

  const unsigned gM = (2 * (M / 16)) / WAVES_PER_BLOCK;      // 64  (fc1)
  const unsigned gE = (2 * (E / 16)) / WAVES_PER_BLOCK;      // 16  (fc2)
  const unsigned gH = (2 * (2 * E / 16)) / WAVES_PER_BLOCK;  // 32  (heads)
  const unsigned gL = (4 * (V / 16)) / WAVES_PER_BLOCK;      // 128 (logits)
  const size_t ldsBytes = (size_t)64 * E * 2;                // 128KB

  for (int t = 0; t < T; ++t) {
    step_init<<<dim3(E / TPB, B), TPB, 0, stream>>>(tokens, t, T, emb_w, state, inp, h, E);
    for (int l = 0; l < L; ++l) {
      rms_bf16<<<B, TPB, 0, stream>>>(h, nullptr, bnw + (size_t)l * E, xn_bf, E);
      gemm_nt<1, 0><<<gM, TPB, 0, stream>>>(xn_bf, fc1_bf + (size_t)l * M * E,
                                            fc1b + (size_t)l * M, nullptr,
                                            nullptr, y_bf, E, M, M);
      gemm_nt<0, 1><<<gE, TPB, 0, stream>>>(y_bf, fc2_bf + (size_t)l * E * M,
                                            fc2b + (size_t)l * E, h, h, nullptr,
                                            M, E, E);
    }
    rms_bf16<<<B, TPB, 0, stream>>>(inp, h, normw, xn_bf, E);
    head_gemm<<<gH, TPB, 0, stream>>>(xn_bf, heads_bf, curb, nxtb,
                                      cur_emb, state, emb_cat_bf, E);
    aux_kernel<<<1, TPB, 0, stream>>>(cur_emb, inp, auxbuf, t, B * E);
    logit_gemm<<<gL, TPB, ldsBytes, stream>>>(emb_cat_bf, emb_bf, cur_logit,
                                              out + (size_t)t * V,
                                              (long long)T * V, E, V);
    ce_kernel<<<dim3(B, 2), TPB, 0, stream>>>(cur_logit, out + (size_t)t * V,
                                              (long long)T * V, tokens, targets,
                                              t, T, V, ce_rows);
  }
  final_reduce<<<1, TPB, 0, stream>>>(ce_rows, auxbuf, out + (size_t)B * T * V);
}
